// MultiHeadAttentionMDAM_50448685859366
// MI455X (gfx1250) — matrix operand order repaired and checked by reading, not compiled, not run on hardware
//
#include <hip/hip_runtime.h>
#include <hip/hip_bf16.h>

// ---------------------------------------------------------------------------
// MHA (E=128, H=8, B=16, S=512) for gfx1250, fused flash-attention style.
// bf16 WMMA 16x16x32 with fp32 accumulation; TDM (tensor_load_to_lds) staging
// of K / V^T tiles with s_wait_tensorcnt; LDS-staged coalesced b128 epilogues.
// ---------------------------------------------------------------------------

typedef __bf16 bf16_t;
typedef __attribute__((ext_vector_type(16))) __bf16 v16bf;
typedef __attribute__((ext_vector_type(8)))  __bf16 v8bf;
typedef __attribute__((ext_vector_type(8)))  float  v8f;
typedef __attribute__((ext_vector_type(4)))  unsigned int u32x4;
typedef __attribute__((ext_vector_type(8)))  int i32x8;
typedef __attribute__((ext_vector_type(4)))  int i32x4;

constexpr int kE  = 128;   // embed dim (= head dim)
constexpr int kH  = 8;     // heads
constexpr int kB  = 16;    // batch
constexpr int kS  = 512;   // sequence (graph) length

// ---- TDM availability ------------------------------------------------------
#if defined(__has_builtin)
#if __has_builtin(__builtin_amdgcn_tensor_load_to_lds)
#define USE_TDM 1
#endif
#endif
#ifndef USE_TDM
#define USE_TDM 0
#endif
#if __has_include(<hip/amd_detail/amd_gfx1250_TDM.h>)
#define TDM_6ARG 1
#else
#define TDM_6ARG 0
#endif

__device__ __forceinline__ v8f wmma_bf16(v16bf a, v16bf b, v8f c) {
  // (neg_a, A, neg_b, B, c_mod, C, reuse_a, reuse_b)
  return __builtin_amdgcn_wmma_f32_16x16x32_bf16(false, a, false, b,
                                                 (short)0, c, false, false);
}

// A-fragment (16x32 bf16, MxK) from a row-major tile (stride in elements).
// Lane ln: m = ln&15, kh = ln>>4.  Elems e -> K = kh*16 + e
// (contiguous order, matching load_b).
__device__ __forceinline__ v16bf load_a(const bf16_t* p, int stride, int lane) {
  int m = lane & 15, kh = lane >> 4;
  const bf16_t* r0 = p + m * stride + kh * 16;
  v16bf a;
  ((v8bf*)&a)[0] = *(const v8bf*)(r0);
  ((v8bf*)&a)[1] = *(const v8bf*)(r0 + 8);
  return a;
}

// B-fragment (32x16 bf16, KxN) from an n-major ("B-transposed") tile:
// bt[n][k], stride in elements. Lane ln: n = ln&15, kh = ln>>4,
// elems e -> K = kh*16 + e  => one contiguous 32-byte load.
__device__ __forceinline__ v16bf load_b(const bf16_t* bt, int stride, int lane) {
  int n = lane & 15, kh = lane >> 4;
  return *(const v16bf*)(bt + n * stride + kh * 16);
}

#if USE_TDM
// Issue a 2-D TDM tile load (global -> LDS), bf16 elements.
// D# per ISA 8.3/8.4: group0 = {count=1 | lds_addr | global_addr | type=2},
// group1 = {data_size=2B, tensor_dim0/1, tile_dim0/1, tensor_dim0_stride}.
__device__ __forceinline__ void tdm_load_2d(void* ldsPtr, const void* gPtr,
                                            unsigned tdim0, unsigned tdim1,
                                            unsigned tile0, unsigned tile1,
                                            unsigned stride0) {
  unsigned long long ga = (unsigned long long)(size_t)gPtr;
  unsigned ldsOff = (unsigned)(size_t)ldsPtr;   // flat addr[31:0] == LDS offset
  u32x4 g0;
  g0[0] = 1u;                                           // count=1, user D#
  g0[1] = ldsOff;                                       // lds_addr
  g0[2] = (unsigned)ga;                                 // global_addr[31:0]
  g0[3] = (unsigned)((ga >> 32) & 0x1FFFFFFu) | (2u << 30);  // [56:32] | type=2
  i32x8 g1;
  g1[0] = (int)(1u << 16);                              // data_size = 2 bytes
  g1[1] = (int)((tdim0 & 0xFFFFu) << 16);               // tensor_dim0[15:0]
  g1[2] = (int)((tdim0 >> 16) | ((tdim1 & 0xFFFFu) << 16));
  g1[3] = (int)((tdim1 >> 16) | (tile0 << 16));         // tile_dim0
  g1[4] = (int)tile1;                                   // tile_dim1 (tile_dim2=0)
  g1[5] = (int)stride0;                                 // dim0 stride (elems)
  g1[6] = 0;
  g1[7] = 0;
  i32x4 z = {0, 0, 0, 0};                               // groups 2/3: 2-D tensor
#if TDM_6ARG
  i32x8 z8 = {0, 0, 0, 0, 0, 0, 0, 0};
  __builtin_amdgcn_tensor_load_to_lds(g0, g1, z, z, z8, 0);
#else
  __builtin_amdgcn_tensor_load_to_lds(g0, g1, z, z, 0);
#endif
}

__device__ __forceinline__ void tdm_wait() {
#if __has_builtin(__builtin_amdgcn_s_wait_tensorcnt)
  __builtin_amdgcn_s_wait_tensorcnt((short)0);
#else
  asm volatile("s_wait_tensorcnt 0x0" ::: "memory");
#endif
}
#endif  // USE_TDM

// ===========================================================================
// Kernel 1: per-(h,b) projections.  Q,K row-major [hb][s][e]; V transposed
// [hb][e][s] so the PV GEMM can use it directly as an n-major B operand.
// C-tiles are staged per-wave in LDS and flushed as coalesced b128 stores.
// ===========================================================================
__global__ void proj_kernel(const float* __restrict__ q,
                            const float* __restrict__ Wq,
                            const float* __restrict__ Wk,
                            const float* __restrict__ Wv,
                            bf16_t* __restrict__ Qp,
                            bf16_t* __restrict__ Kp,
                            bf16_t* __restrict__ Vt) {
  extern __shared__ char smem[];
  bf16_t* wqT = (bf16_t*)smem;            // [128 n][128 k]  (Wq transposed)
  bf16_t* wkT = wqT + kE * kE;            // [128 n][128 k]
  bf16_t* wvT = wkT + kE * kE;            // [128 e][128 k]  (A operand for V^T)
  bf16_t* xs  = wvT + kE * kE;            // [128 s][128 k]  current X block
  bf16_t* stgAll = xs + 128 * kE;         // [8 wave][16][128] epilogue staging

  const int wg = blockIdx.x;              // h*16 + b
  const int h = wg >> 4, b = wg & 15;
  const int tid = threadIdx.x, lane = tid & 31, wave = tid >> 5;
  const int hb = h * kB + b;
  bf16_t* stg = stgAll + wave * 16 * kE;

  const float* wq = Wq + (size_t)h * kE * kE;
  const float* wk = Wk + (size_t)h * kE * kE;
  const float* wv = Wv + (size_t)h * kE * kE;
  for (int i = tid; i < kE * kE; i += 256) {
    int kin = i >> 7, nout = i & 127;     // coalesced read W[k][n]
    wqT[nout * kE + kin] = (bf16_t)wq[i];
    wkT[nout * kE + kin] = (bf16_t)wk[i];
    wvT[nout * kE + kin] = (bf16_t)wv[i];
  }

  const size_t qkBase = (size_t)hb * kS * kE;   // Q/K row-major base
  const size_t vtBase = (size_t)hb * kE * kS;   // V^T base
  const int mloc = (lane >> 4) * 8, n = lane & 15;

  for (int it = 0; it < 4; ++it) {              // 4 blocks of 128 rows
    __syncthreads();
    const float* xsrc = q + ((size_t)b * kS + it * 128) * kE;
    if (it < 3) __builtin_prefetch(xsrc + 128 * kE, 0, 1);
    for (int i = tid; i < 128 * kE; i += 256) xs[i] = (bf16_t)xsrc[i];
    __syncthreads();

    const int m0 = wave * 16;                   // wave's 16-row stripe

    // ---- Q then K stripe: [16 x 128] = X_stripe @ W -----------------------
    for (int t = 0; t < 2; ++t) {
      const bf16_t* wT = t ? wkT : wqT;
      bf16_t* dst = t ? Kp : Qp;
      #pragma unroll
      for (int nt = 0; nt < 8; ++nt) {
        v8f acc = {0.f,0.f,0.f,0.f,0.f,0.f,0.f,0.f};
        #pragma unroll
        for (int kc = 0; kc < 4; ++kc) {
          v16bf a = load_a(xs + m0 * kE + kc * 32, kE, lane);
          acc = wmma_bf16(a, load_b(wT + nt * 16 * kE + kc * 32, kE, lane), acc);
        }
        #pragma unroll
        for (int r = 0; r < 8; ++r)
          stg[(mloc + r) * kE + nt * 16 + n] = (bf16_t)acc[r];
      }
      asm volatile("s_wait_dscnt 0x0" ::: "memory");   // wave-local DS RAW
      const uint4* s4 = (const uint4*)stg;
      #pragma unroll
      for (int i = lane; i < 256; i += 32) {           // 16 rows x 16 uint4
        int m = i >> 4, c = i & 15;
        *((uint4*)(dst + qkBase + (size_t)(it * 128 + m0 + m) * kE) + c) = s4[i];
      }
    }

    // ---- V^T stripe: Vt[e][s] = (Wv^T @ X^T); A=wvT rows, B=xs rows -------
    #pragma unroll
    for (int nt = 0; nt < 8; ++nt) {
      v8f acc = {0.f,0.f,0.f,0.f,0.f,0.f,0.f,0.f};
      #pragma unroll
      for (int kc = 0; kc < 4; ++kc) {
        v16bf a = load_a(wvT + m0 * kE + kc * 32, kE, lane);
        acc = wmma_bf16(a, load_b(xs + nt * 16 * kE + kc * 32, kE, lane), acc);
      }
      #pragma unroll
      for (int r = 0; r < 8; ++r)
        stg[(mloc + r) * 128 + nt * 16 + n] = (bf16_t)acc[r];  // [16 e][128 s]
    }
    asm volatile("s_wait_dscnt 0x0" ::: "memory");
    const uint4* s4 = (const uint4*)stg;
    #pragma unroll
    for (int i = lane; i < 256; i += 32) {
      int m = i >> 4, c = i & 15;          // e-row m, 16B chunk c of 128 s-cols
      *((uint4*)(Vt + vtBase + (size_t)(m0 + m) * kS + it * 128) + c) = s4[i];
    }
  }
}

// ===========================================================================
// Kernel 2: fused masked flash attention per (h, b, 128-q-row block).
// 8 waves x 16 q-rows; KV streamed in 64-key blocks into LDS via TDM.
// ===========================================================================
__global__ void attn_kernel(const bf16_t* __restrict__ Qp,
                            const bf16_t* __restrict__ Kp,
                            const bf16_t* __restrict__ Vt,
                            const unsigned char* __restrict__ mask,
                            bf16_t* __restrict__ Hp) {
  extern __shared__ char smem[];
  bf16_t* ks = (bf16_t*)smem;            // [64 key][128 e]   16 KB
  bf16_t* vs = ks + 64 * kE;             // [128 e][64 key]   16 KB
  bf16_t* ps = vs + kE * 64;             // [8 wave][16][64]  16 KB

  const int wg = blockIdx.x;             // h*64 + b*4 + qb
  const int qb = wg & 3, b = (wg >> 2) & 15, h = wg >> 6;
  const int tid = threadIdx.x, lane = tid & 31, wave = tid >> 5;
  const int hb = h * kB + b;
  const size_t qkBase = (size_t)hb * kS * kE;
  const size_t vtBase = (size_t)hb * kE * kS;
  const int q0 = qb * 128 + wave * 16;
  const int mloc = (lane >> 4) * 8, n = lane & 15;
  const float nf = 0.08838834764831845f;       // 1/sqrt(128)

  // Preload this wave's Q stripe as 4 A-fragments (16 x 128 bf16).
  v16bf qa[4];
  #pragma unroll
  for (int kc = 0; kc < 4; ++kc)
    qa[kc] = load_a(Qp + qkBase + (size_t)q0 * kE + kc * 32, kE, lane);

  float row_max[8], row_sum[8];
  v8f o[8];
  #pragma unroll
  for (int r = 0; r < 8; ++r) { row_max[r] = -3.0e38f; row_sum[r] = 0.f; }
  #pragma unroll
  for (int t = 0; t < 8; ++t) o[t] = (v8f){0.f,0.f,0.f,0.f,0.f,0.f,0.f,0.f};

  for (int kb = 0; kb < kS; kb += 64) {
    __syncthreads();                     // all waves done reading ks/vs
    if (kb + 64 < kS)                    // global_prefetch_b8 of next K block
      __builtin_prefetch(Kp + qkBase + (size_t)(kb + 64) * kE, 0, 1);
#if USE_TDM
    if (wave == 0) {                     // TDM DMA: one issue per WG
      tdm_load_2d(ks, Kp + qkBase + (size_t)kb * kE,
                  /*tdim0=*/kE, /*tdim1=*/kS, /*tile0=*/kE, /*tile1=*/64,
                  /*stride0=*/kE);
      tdm_load_2d(vs, Vt + vtBase + kb,
                  /*tdim0=*/kS, /*tdim1=*/kE, /*tile0=*/64, /*tile1=*/kE,
                  /*stride0=*/kS);
      tdm_wait();
    }
#else
    { // cooperative staging fallback
      const uint4* src = (const uint4*)(Kp + qkBase + (size_t)kb * kE);
      uint4* dst = (uint4*)ks;
      for (int i = tid; i < 64 * kE / 8; i += 256) dst[i] = src[i];
      for (int i = tid; i < kE * 64 / 8; i += 256) {
        int e = i >> 3, c = i & 7;       // 8 x 16B chunks per 64-key row
        ((uint4*)vs)[i] = *(const uint4*)(Vt + vtBase + (size_t)e * kS + kb + c * 8);
      }
    }
#endif
    __syncthreads();

    // ---- S = nf * Q K^T  (16 x 64, fp32 accum) ----------------------------
    v8f s[4];
    #pragma unroll
    for (int nt = 0; nt < 4; ++nt) {
      v8f acc = {0.f,0.f,0.f,0.f,0.f,0.f,0.f,0.f};
      #pragma unroll
      for (int kc = 0; kc < 4; ++kc)
        acc = wmma_bf16(qa[kc], load_b(ks + nt * 16 * kE + kc * 32, kE, lane), acc);
      s[nt] = acc;
    }

    // ---- scale, mask, online softmax --------------------------------------
    const unsigned char* mrow = mask + ((size_t)b * kS + q0) * kS + kb;
    #pragma unroll
    for (int r = 0; r < 8; ++r) {
      int m = mloc + r;
      float mx = -3.0e38f;
      #pragma unroll
      for (int nt = 0; nt < 4; ++nt) {
        float v = s[nt][r] * nf;
        if (mrow[(size_t)m * kS + nt * 16 + n]) v = -3.0e38f;  // -> exp ~ 0
        s[nt][r] = v;
        mx = fmaxf(mx, v);
      }
      #pragma unroll
      for (int off = 1; off < 16; off <<= 1)      // reduce across 16-lane half
        mx = fmaxf(mx, __shfl_xor(mx, off, 32));

      float nm = fmaxf(row_max[r], mx);
      float scale = __expf(row_max[r] - nm);
      row_max[r] = nm;
      float rs = 0.f;
      #pragma unroll
      for (int nt = 0; nt < 4; ++nt) {
        float p = __expf(s[nt][r] - nm);
        s[nt][r] = p;
        rs += p;
      }
      #pragma unroll
      for (int off = 1; off < 16; off <<= 1)
        rs += __shfl_xor(rs, off, 32);
      row_sum[r] = row_sum[r] * scale + rs;
      #pragma unroll
      for (int t = 0; t < 8; ++t) o[t][r] *= scale;   // rescale running O
    }

    // ---- C-layout -> A-layout via LDS (wave-private region) ---------------
    bf16_t* pw = ps + wave * 16 * 64;
    #pragma unroll
    for (int r = 0; r < 8; ++r) {
      int m = mloc + r;
      #pragma unroll
      for (int nt = 0; nt < 4; ++nt)
        pw[m * 64 + nt * 16 + n] = (bf16_t)s[nt][r];
    }
    asm volatile("s_wait_dscnt 0x0" ::: "memory");   // wave-local DS RAW

    // ---- O += P @ V  (B operand = vs, n-major over e) ----------------------
    #pragma unroll
    for (int kc = 0; kc < 2; ++kc) {
      v16bf pa = load_a(pw + kc * 32, 64, lane);
      #pragma unroll
      for (int ont = 0; ont < 8; ++ont)
        o[ont] = wmma_bf16(pa, load_b(vs + ont * 16 * 64 + kc * 32, 64, lane), o[ont]);
    }
  }

  // ---- normalize; stage in LDS (reuse ks/vs region); coalesced flush -------
  __syncthreads();                          // ks/vs dead for all waves
  bf16_t* hstg = ((bf16_t*)smem) + wave * 16 * kE;   // 4 KB per wave
  #pragma unroll
  for (int r = 0; r < 8; ++r) {
    int m = mloc + r;
    float inv = 1.f / row_sum[r];
    #pragma unroll
    for (int ont = 0; ont < 8; ++ont)
      hstg[m * kE + ont * 16 + n] = (bf16_t)(o[ont][r] * inv);
  }
  asm volatile("s_wait_dscnt 0x0" ::: "memory");
  const uint4* s4 = (const uint4*)hstg;
  #pragma unroll
  for (int i = lane; i < 256; i += 32) {    // 16 rows x 16 uint4
    int m = i >> 4, c = i & 15;
    *((uint4*)(Hp + qkBase + (size_t)(q0 + m) * kE) + c) = s4[i];
  }
}

// ===========================================================================
// Kernel 3: out[b,q,:] = sum_h heads[h,b,q,:] @ Wout[h]   (fp32 output)
// ===========================================================================
__global__ void out_kernel(const bf16_t* __restrict__ Hp,
                           const float* __restrict__ Wo,
                           float* __restrict__ out) {
  extern __shared__ char smem[];
  bf16_t* wt = (bf16_t*)smem;            // [128 d][128 e]  current head WoT

  const int wg = blockIdx.x;             // b*4 + qb
  const int qb = wg & 3, b = wg >> 2;
  const int tid = threadIdx.x, lane = tid & 31, wave = tid >> 5;
  const int q0 = qb * 128 + wave * 16;
  const int mloc = (lane >> 4) * 8, n = lane & 15;

  v8f o[8];
  #pragma unroll
  for (int t = 0; t < 8; ++t) o[t] = (v8f){0.f,0.f,0.f,0.f,0.f,0.f,0.f,0.f};

  for (int h = 0; h < kH; ++h) {
    __syncthreads();
    const float* wsrc = Wo + (size_t)h * kE * kE;     // [e][d]
    for (int i = tid; i < kE * kE; i += 256) {
      int e = i >> 7, d = i & 127;
      wt[d * kE + e] = (bf16_t)wsrc[i];
    }
    __syncthreads();

    const size_t hbase = ((size_t)(h * kB + b) * kS + q0) * kE;
    v16bf a[4];
    #pragma unroll
    for (int kc = 0; kc < 4; ++kc)
      a[kc] = load_a(Hp + hbase + kc * 32, kE, lane);
    #pragma unroll
    for (int nt = 0; nt < 8; ++nt)
      #pragma unroll
      for (int kc = 0; kc < 4; ++kc)
        o[nt] = wmma_bf16(a[kc], load_b(wt + nt * 16 * kE + kc * 32, kE, lane), o[nt]);
  }

  // f32 b32 stores: lanes 0-15 cover 16 consecutive dwords -> coalesced
  #pragma unroll
  for (int r = 0; r < 8; ++r)
    #pragma unroll
    for (int nt = 0; nt < 8; ++nt)
      out[((size_t)b * kS + q0 + mloc + r) * kE + nt * 16 + n] = o[nt][r];
}

// ===========================================================================
extern "C" void kernel_launch(void* const* d_in, const int* in_sizes, int n_in,
                              void* d_out, int out_size, void* d_ws, size_t ws_size,
                              hipStream_t stream) {
  const float*         q    = (const float*)d_in[0];
  const unsigned char* mask = (const unsigned char*)d_in[1];  // jnp.bool_ = 1 byte
  const float*         Wq   = (const float*)d_in[2];
  const float*         Wk   = (const float*)d_in[3];
  const float*         Wv   = (const float*)d_in[4];
  const float*         Wo   = (const float*)d_in[5];

  const size_t perTensor = (size_t)kH * kB * kS * kE;   // 8.4M bf16 elems
  bf16_t* Qp = (bf16_t*)d_ws;
  bf16_t* Kp = Qp + perTensor;
  bf16_t* Vt = Kp + perTensor;
  bf16_t* Hp = Vt + perTensor;

  // Kernel 1: 128 WGs (one per (h,b)), 8 waves; 160 KB dynamic LDS.
  proj_kernel<<<dim3(kH * kB), dim3(256),
                (4 * kE * kE + 8 * 16 * kE) * sizeof(bf16_t), stream>>>(
      q, Wq, Wk, Wv, Qp, Kp, Vt);

  // Kernel 2: 512 WGs (h,b,q-block); 48 KB LDS.
  attn_kernel<<<dim3(kH * kB * 4), dim3(256), 48 * 1024, stream>>>(
      Qp, Kp, Vt, mask, Hp);

  // Kernel 3: 64 WGs (b,q-block); 32 KB LDS.
  out_kernel<<<dim3(kB * 4), dim3(256), kE * kE * sizeof(bf16_t), stream>>>(
      Hp, Wo, (float*)d_out);
}
